// CustomExpert_35983236005997
// MI455X (gfx1250) — matrix operand order repaired and checked by reading, not compiled, not run on hardware
//
#include <hip/hip_runtime.h>

// MoE grouped-FFN: FC1 -> ReLU -> FC2, fused per 16-token wave tile.
// E=64 experts, 1024 tokens/expert, D_IN=1024, H=64, D_OUT=1024.
// f32 in/out, f16 WMMA with f32 accumulation (memory-bound regime on MI455X).
// Weights are pre-converted to f16 in d_ws by a cheap pre-pass so the hot
// loops load packed f16 fragments directly (no per-wave v_cvt on weights).

typedef __attribute__((ext_vector_type(16))) _Float16 v16h;
typedef __attribute__((ext_vector_type(8)))  _Float16 v8h;
typedef __attribute__((ext_vector_type(8)))  float    v8f;
typedef __attribute__((ext_vector_type(4)))  float    v4f;

#define NEXP   64
#define HID    64
#define DIN    1024
#define DOUT   1024
#define PER    1024          // tokens per expert
#define WAVES  8             // waves per block
#define HSTR   80            // LDS row stride in halves (160B, 16B-aligned)

#define W1ELEMS (NEXP * HID * DIN)    // 4,194,304
#define W2ELEMS (NEXP * DOUT * HID)   // 4,194,304

// Convert 8 consecutive f32 -> 8 f16 (two aligned b128 loads + packed cvts).
__device__ __forceinline__ v8h cvt8(const float* __restrict__ p) {
    v4f a = *(const v4f*)p;
    v4f b = *(const v4f*)(p + 4);
    v8h r;
    r[0] = (_Float16)a[0]; r[1] = (_Float16)a[1];
    r[2] = (_Float16)a[2]; r[3] = (_Float16)a[3];
    r[4] = (_Float16)b[0]; r[5] = (_Float16)b[1];
    r[6] = (_Float16)b[2]; r[7] = (_Float16)b[3];
    return r;
}

__device__ __forceinline__ v16h join16(v8h lo, v8h hi) {
    return __builtin_shufflevector(lo, hi, 0,1,2,3,4,5,6,7,8,9,10,11,12,13,14,15);
}

// ---------------------------------------------------------------------------
// Pre-pass: f32 -> f16 weight conversion (one v8h store per thread-iter).
// ---------------------------------------------------------------------------
__global__ __launch_bounds__(256) void wcvt_f16(const float* __restrict__ in,
                                                _Float16* __restrict__ out,
                                                int n8) {
    int i = blockIdx.x * blockDim.x + threadIdx.x;
    if (i < n8) {
        ((v8h*)out)[i] = cvt8(in + 8 * (size_t)i);
    }
}

// ---------------------------------------------------------------------------
// Main fused kernel, f16 weights (fast path).
// ---------------------------------------------------------------------------
__global__ __launch_bounds__(256) void moe_ffn_wmma_h(
    const float*    __restrict__ x,    // [T, DIN] f32
    const _Float16* __restrict__ w1,   // [E, HID, DIN] f16
    const _Float16* __restrict__ w2,   // [E, DOUT, HID] f16
    float*          __restrict__ y)    // [T, DOUT] f32
{
    __shared__ _Float16 hbuf[WAVES * 16 * HSTR];

    const int tid  = threadIdx.x;
    const int wave = tid >> 5;
    const int lane = tid & 31;
    const int lrow = lane & 15;
    const int hi   = lane >> 4;        // half-wave selector
    const int koffA = hi ? 8 : 0;      // A-fragment K offset within 16-group
    const int kloB  = hi ? 16 : 0;     // unused; B fragments use koffA

    const int    e    = blockIdx.x >> 3;                        // expert id
    const int    mloc = ((blockIdx.x & 7) * WAVES + wave) * 16; // token tile in expert
    const size_t row0 = (size_t)e * PER + mloc;                 // first global token row

    const float*    __restrict__ xa  = x  + (row0 + lrow) * (size_t)DIN;
    const _Float16* __restrict__ w1e = w1 + (size_t)e * HID * DIN;
    const _Float16* __restrict__ w2e = w2 + (size_t)e * DOUT * HID;

    // ---------------- FC1: h[16,64] = x_tile[16,1024] @ W1e^T ----------------
    v8f acc[4] = {};  // 4 N-tiles of 16x16 f32
    for (int k0 = 0; k0 < DIN; k0 += 32) {
        // A fragment (16x32 f16): lanes0-15 K={0..7,16..23}, lanes16-31 K={8..15,24..31}
        v16h a = join16(cvt8(xa + k0 + koffA), cvt8(xa + k0 + koffA + 16));
        __builtin_prefetch(xa + k0 + 128, 0, 1);  // global_prefetch_b8 next chunks
#pragma unroll
        for (int t = 0; t < 4; ++t) {
            // B fragment (32x16): lane = column n = t*16+lrow; K={0..7,16..23} / {8..15,24..31}
            const _Float16* bp = w1e + (size_t)(t * 16 + lrow) * DIN + k0 + koffA;
            v16h b = join16(*(const v8h*)bp, *(const v8h*)(bp + 16));
            acc[t] = __builtin_amdgcn_wmma_f32_16x16x32_f16(
                false, a, false, b, (short)0, acc[t], false, false);
        }
    }

    // ---------------- ReLU + stage h in LDS as f16 (row-major) ----------------
    _Float16* hw = hbuf + wave * 16 * HSTR;
#pragma unroll
    for (int t = 0; t < 4; ++t) {
#pragma unroll
        for (int v = 0; v < 8; ++v) {
            const int r = hi ? (v + 8) : v;   // C/D layout: lanes16-31 hold M=8..15
            const float z = acc[t][v];
            hw[r * HSTR + t * 16 + lrow] = (_Float16)(z > 0.0f ? z : 0.0f);
        }
    }

    // Re-read h in WMMA A-fragment layout (same wave; DScnt ordering suffices).
    v16h a2[2];
#pragma unroll
    for (int kk = 0; kk < 2; ++kk) {
        const _Float16* p = hw + lrow * HSTR + kk * 32 + koffA;  // 16B-aligned
        a2[kk] = join16(*(const v8h*)p, *(const v8h*)(p + 16));
    }

    // ---------------- FC2: y[16,1024] = relu(h)[16,64] @ W2e^T ----------------
    float* __restrict__ yo = y + row0 * (size_t)DOUT;
    for (int nb = 0; nb < DOUT; nb += 128) {
        v8f c2[8] = {};
#pragma unroll
        for (int t = 0; t < 8; ++t) {
            const _Float16* bp = w2e + (size_t)(nb + t * 16 + lrow) * HID + koffA;
#pragma unroll
            for (int kk = 0; kk < 2; ++kk) {
                v16h b = join16(*(const v8h*)(bp + kk * 32),
                                *(const v8h*)(bp + kk * 32 + 16));
                c2[t] = __builtin_amdgcn_wmma_f32_16x16x32_f16(
                    false, a2[kk], false, b, (short)0, c2[t], false, false);
            }
        }
#pragma unroll
        for (int t = 0; t < 8; ++t) {
#pragma unroll
            for (int v = 0; v < 8; ++v) {
                const int r = hi ? (v + 8) : v;
                yo[(size_t)r * DOUT + nb + t * 16 + lrow] = c2[t][v];
            }
        }
    }
}

// ---------------------------------------------------------------------------
// Fallback fused kernel, f32 weights converted in-register (if d_ws too small).
// ---------------------------------------------------------------------------
__global__ __launch_bounds__(256) void moe_ffn_wmma_f(
    const float* __restrict__ x,
    const float* __restrict__ w1,
    const float* __restrict__ w2,
    float* __restrict__ y)
{
    __shared__ _Float16 hbuf[WAVES * 16 * HSTR];

    const int tid  = threadIdx.x;
    const int wave = tid >> 5;
    const int lane = tid & 31;
    const int lrow = lane & 15;
    const int hi   = lane >> 4;
    const int koffA = hi ? 8 : 0;
    const int kloB  = hi ? 16 : 0;

    const int    e    = blockIdx.x >> 3;
    const int    mloc = ((blockIdx.x & 7) * WAVES + wave) * 16;
    const size_t row0 = (size_t)e * PER + mloc;

    const float* __restrict__ xa  = x  + (row0 + lrow) * (size_t)DIN;
    const float* __restrict__ w1e = w1 + (size_t)e * HID * DIN;
    const float* __restrict__ w2e = w2 + (size_t)e * DOUT * HID;

    v8f acc[4] = {};
    for (int k0 = 0; k0 < DIN; k0 += 32) {
        v16h a = join16(cvt8(xa + k0 + koffA), cvt8(xa + k0 + koffA + 16));
#pragma unroll
        for (int t = 0; t < 4; ++t) {
            const float* bp = w1e + (size_t)(t * 16 + lrow) * DIN + k0 + koffA;
            v16h b = join16(cvt8(bp), cvt8(bp + 16));
            acc[t] = __builtin_amdgcn_wmma_f32_16x16x32_f16(
                false, a, false, b, (short)0, acc[t], false, false);
        }
    }

    _Float16* hw = hbuf + wave * 16 * HSTR;
#pragma unroll
    for (int t = 0; t < 4; ++t) {
#pragma unroll
        for (int v = 0; v < 8; ++v) {
            const int r = hi ? (v + 8) : v;
            const float z = acc[t][v];
            hw[r * HSTR + t * 16 + lrow] = (_Float16)(z > 0.0f ? z : 0.0f);
        }
    }

    v16h a2[2];
#pragma unroll
    for (int kk = 0; kk < 2; ++kk) {
        const _Float16* p = hw + lrow * HSTR + kk * 32 + koffA;
        a2[kk] = join16(*(const v8h*)p, *(const v8h*)(p + 16));
    }

    float* __restrict__ yo = y + row0 * (size_t)DOUT;
    for (int nb = 0; nb < DOUT; nb += 128) {
        v8f c2[8] = {};
#pragma unroll
        for (int t = 0; t < 8; ++t) {
            const float* bp = w2e + (size_t)(nb + t * 16 + lrow) * HID + koffA;
#pragma unroll
            for (int kk = 0; kk < 2; ++kk) {
                v16h b = join16(cvt8(bp + kk * 32), cvt8(bp + kk * 32 + 16));
                c2[t] = __builtin_amdgcn_wmma_f32_16x16x32_f16(
                    false, a2[kk], false, b, (short)0, c2[t], false, false);
            }
        }
#pragma unroll
        for (int t = 0; t < 8; ++t) {
#pragma unroll
            for (int v = 0; v < 8; ++v) {
                const int r = hi ? (v + 8) : v;
                yo[(size_t)r * DOUT + nb + t * 16 + lrow] = c2[t][v];
            }
        }
    }
}

extern "C" void kernel_launch(void* const* d_in, const int* in_sizes, int n_in,
                              void* d_out, int out_size, void* d_ws, size_t ws_size,
                              hipStream_t stream) {
    (void)in_sizes; (void)n_in; (void)out_size;
    const float* x  = (const float*)d_in[0];
    // d_in[1] = fwd_expert_count (equal groups; layout fixed, unused)
    const float* w1 = (const float*)d_in[2];
    const float* w2 = (const float*)d_in[3];
    float* y = (float*)d_out;

    const size_t need = (size_t)(W1ELEMS + W2ELEMS) * sizeof(_Float16); // 16.8 MB
    dim3 grid(512), block(256);  // T/16 = 4096 wave tiles, 8 waves/block

    if (d_ws != nullptr && ws_size >= need) {
        // Deterministic pre-pass every call: weights -> f16 in workspace.
        _Float16* w1h = (_Float16*)d_ws;
        _Float16* w2h = w1h + W1ELEMS;
        hipLaunchKernelGGL(wcvt_f16, dim3(W1ELEMS / 8 / 256), dim3(256), 0, stream,
                           w1, w1h, W1ELEMS / 8);
        hipLaunchKernelGGL(wcvt_f16, dim3(W2ELEMS / 8 / 256), dim3(256), 0, stream,
                           w2, w2h, W2ELEMS / 8);
        hipLaunchKernelGGL(moe_ffn_wmma_h, grid, block, 0, stream, x, w1h, w2h, y);
    } else {
        hipLaunchKernelGGL(moe_ffn_wmma_f, grid, block, 0, stream, x, w1, w2, y);
    }
}
